// Model_19550691131546
// MI455X (gfx1250) — matrix operand order repaired and checked by reading, not compiled, not run on hardware
//
#include <hip/hip_runtime.h>
#include <hip/hip_bf16.h>

// Problem constants (match reference)
#define B_    4
#define L_    8192
#define DIN   64
#define H_    256
#define N2_   32
#define NL_   4
#define DOUT  10
#define NC_   8               // chunks for the parallel scan
#define TCH   (L_ / NC_)      // 1024 steps per chunk

typedef __bf16 bf16;
typedef __attribute__((ext_vector_type(16))) __bf16 v16bf;
typedef __attribute__((ext_vector_type(8)))  __bf16 v8bf;
typedef __attribute__((ext_vector_type(8)))  float  v8f;
typedef __attribute__((ext_vector_type(4))) unsigned int u32x4;
typedef __attribute__((ext_vector_type(8))) int         i32x8;
typedef __attribute__((ext_vector_type(4))) int         i32x4;

#if __has_builtin(__builtin_amdgcn_tensor_load_to_lds)
#define HAVE_TDM 1
#else
#define HAVE_TDM 0
#endif

#if HAVE_TDM
// Issue one 2D Tensor-Data-Mover tile load (D# per CDNA5 ISA §8.3/8.4) and
// wait for completion. Caller must be a single wave; TDM ignores EXEC.
__device__ inline void tdm_load_2d(unsigned lds_off, const void* gaddr,
                                   unsigned data_size_log2,   // 0=1B,1=2B,2=4B
                                   unsigned tensor_d0, unsigned tensor_d1,
                                   unsigned tile_d0, unsigned tile_d1,
                                   unsigned stride0) {
  unsigned long long ga = (unsigned long long)(size_t)gaddr;
  u32x4 g0;
  g0[0] = 1u;                                        // count=1 (valid user D#)
  g0[1] = lds_off;                                   // lds_addr   [63:32]
  g0[2] = (unsigned)(ga & 0xffffffffu);              // global_addr[95:64]
  g0[3] = (unsigned)((ga >> 32) & 0x01ffffffu)       // global_addr[120:96]
        | (2u << 30);                                // type=2 ("image")
  i32x8 g1;
  g1[0] = (int)(data_size_log2 << 16);               // wg_mask=0 | data_size
  g1[1] = (int)(tensor_d0 << 16);                    // tensor_dim0 lo16 @ 63:48
  g1[2] = (int)((tensor_d0 >> 16) | (tensor_d1 << 16));
  g1[3] = (int)((tensor_d1 >> 16) | (tile_d0 << 16));
  g1[4] = (int)(tile_d1 & 0xffff);                   // tile_dim1 | tile_dim2=0
  g1[5] = (int)stride0;                              // tensor_dim0_stride lo32
  g1[6] = 0;
  g1[7] = 0;
  i32x4 z4 = {0, 0, 0, 0};
#if defined(__clang_major__) && (__clang_major__ >= 23)
  i32x8 z8 = {0, 0, 0, 0, 0, 0, 0, 0};
  __builtin_amdgcn_tensor_load_to_lds(g0, g1, z4, z4, z8, 0);
#else
  __builtin_amdgcn_tensor_load_to_lds(g0, g1, z4, z4, 0);
#endif
  __builtin_amdgcn_s_wait_tensorcnt(0);
}
#endif

// ---------------------------------------------------------------------------
// 1) Precompute per-mode discretized params: w = exp(dt*A), Ct = C*(w-1)/A
//    wct layout: [NL][H][N2] x float4 {w_re, w_im, ct_re, ct_im}
// ---------------------------------------------------------------------------
__global__ void s4d_params(const float* __restrict__ log_dt,
                           const float* __restrict__ log_A_re,
                           const float* __restrict__ A_im,
                           const float* __restrict__ C_re,
                           const float* __restrict__ C_im,
                           float* __restrict__ wct) {
  int idx = blockIdx.x * blockDim.x + threadIdx.x;
  if (idx >= NL_ * H_ * N2_) return;
  int lh = idx / N2_;                      // (layer, h) flat
  float dt  = __expf(log_dt[lh]);
  float are = -__expf(log_A_re[idx]);
  float aim = A_im[idx];
  float dre = dt * are, dim = dt * aim;
  float e   = __expf(dre);
  float wre = e * __cosf(dim);
  float wim = e * __sinf(dim);
  float m1re = wre - 1.0f, m1im = wim;     // exp(dtA) - 1
  float cre = C_re[idx], cim = C_im[idx];
  float tre = cre * m1re - cim * m1im;     // C * (exp(dtA)-1)
  float tim = cre * m1im + cim * m1re;
  float den = are * are + aim * aim;       // divide by A (mul by conj/|A|^2)
  float ctre = (tre * are + tim * aim) / den;
  float ctim = (tim * are - tre * aim) / den;
  ((float4*)wct)[idx] = make_float4(wre, wim, ctre, ctim);
}

// ---------------------------------------------------------------------------
// 2) fp32 -> bf16 weight conversion (weights stay hot in L2)
// ---------------------------------------------------------------------------
__global__ void f32_to_bf16(const float* __restrict__ in, bf16* __restrict__ out, int n) {
  int i = blockIdx.x * blockDim.x + threadIdx.x;
  if (i < n) out[i] = (bf16)in[i];
}

// ---------------------------------------------------------------------------
// 3) Encoder GEMM: h[b][m][l] = sum_d enc_W[m][d]*x[b][l][d] + enc_b[m]
//    x tile staged by TDM (raw 64x64 fp32), converted to bf16 in LDS.
// ---------------------------------------------------------------------------
__global__ __launch_bounds__(256) void encoder_gemm(
    const float* __restrict__ x,       // (B,L,DIN)
    const bf16*  __restrict__ Wb,      // (H,DIN) bf16
    const float* __restrict__ enc_b,   // (H)
    float*       __restrict__ h)       // (B,H,L)
{
  const int COLS = 64;
#if HAVE_TDM
  __shared__ alignas(16) float xraw[COLS][DIN];   // TDM destination (16 KB)
#endif
  __shared__ alignas(32) bf16 xt[COLS][80];       // [col][d], padded rows
  int tid  = threadIdx.x;
  int col0 = blockIdx.x * COLS;                   // flat (b*L + l)
  int b    = col0 / L_;
  int l0   = col0 % L_;
#if HAVE_TDM
  if (tid < 32)
    tdm_load_2d((unsigned)(size_t)&xraw[0][0],
                x + ((size_t)b * L_ + l0) * DIN,
                /*4B*/2, DIN, COLS, DIN, COLS, DIN);
  __syncthreads();
  for (int i = tid; i < COLS * DIN; i += 256) {
    int c = i >> 6, d = i & 63;
    xt[c][d] = (bf16)xraw[c][d];
  }
#else
  for (int i = tid; i < COLS * DIN; i += 256) {
    int c = i >> 6, d = i & 63;
    xt[c][d] = (bf16)x[((size_t)b * L_ + (l0 + c)) * DIN + d];
  }
#endif
  __syncthreads();

  int wave = tid >> 5, lane = tid & 31;
  int hf = lane >> 4, l16 = lane & 15;
  v8f acc[2][4] = {};                         // 2 row tiles x 4 col tiles
  for (int kk = 0; kk < DIN; kk += 32) {
    v16bf Bf[4];
    for (int ct = 0; ct < 4; ++ct)
      Bf[ct] = *(const v16bf*)&xt[ct * 16 + l16][kk + 16 * hf];
    for (int rt = 0; rt < 2; ++rt) {
      int m = wave * 32 + rt * 16 + l16;
      const bf16* Ar = Wb + (size_t)m * DIN + kk;
      v8bf a0 = *(const v8bf*)(Ar + hf * 16);
      v8bf a1 = *(const v8bf*)(Ar + hf * 16 + 8);
      v16bf A;
      #pragma unroll
      for (int j = 0; j < 8; ++j) { A[j] = a0[j]; A[j + 8] = a1[j]; }
      for (int ct = 0; ct < 4; ++ct)
        acc[rt][ct] = __builtin_amdgcn_wmma_f32_16x16x32_bf16(
            false, A, false, Bf[ct], (short)0, acc[rt][ct], false, false);
    }
  }
  for (int rt = 0; rt < 2; ++rt)
    for (int ct = 0; ct < 4; ++ct)
      #pragma unroll
      for (int v = 0; v < 8; ++v) {
        int row = wave * 32 + rt * 16 + v + 8 * hf;
        int col = ct * 16 + l16;
        h[((size_t)b * H_ + row) * L_ + l0 + col] = acc[rt][ct][v] + enc_b[row];
      }
}

// ---------------------------------------------------------------------------
// 4) S4D recurrence as a chunk-parallel scan (8 chunks of 1024 steps).
//    One wave per (b, channel, chunk); one complex mode per lane.
// ---------------------------------------------------------------------------
__device__ inline float gelu_tanh(float v) {
  const float k0 = 0.7978845608028654f;   // sqrt(2/pi)
  float t = tanhf(k0 * (v + 0.044715f * v * v * v));
  return 0.5f * v * (1.0f + t);
}

// Pass 1: chunk-local end states from zero init (no reduce, no output)
__global__ __launch_bounds__(256) void s4_scan_local(
    const float* __restrict__ h,     // (B,H,L)
    const float* __restrict__ wct,   // layer slice: (H,N2) float4
    float2*      __restrict__ E)     // [B*H][NC][N2]
{
  int wave = threadIdx.x >> 5, lane = threadIdx.x & 31;
  int gidx = blockIdx.x * 8 + wave;            // 0..B*H*NC-1
  int c  = gidx % NC_;
  int bh = gidx / NC_;
  int b = bh / H_, hh = bh % H_;
  float4 p = ((const float4*)wct)[hh * N2_ + lane];
  float wre = p.x, wim = p.y;
  const float* hrow = h + ((size_t)b * H_ + hh) * L_ + (size_t)c * TCH;
  float sre = 0.f, sim = 0.f;
  for (int l0 = 0; l0 < TCH; l0 += 32) {
    float uvec = hrow[l0 + lane];
    #pragma unroll 8
    for (int j = 0; j < 32; ++j) {
      float u = __shfl(uvec, j, 32);
      float nre = wre * sre - wim * sim + u;
      float nim = wre * sim + wim * sre;
      sre = nre; sim = nim;
    }
  }
  E[((size_t)bh * NC_ + c) * N2_ + lane] = make_float2(sre, sim);
}

// Pass 2: tiny sequential scan over 8 chunk states; I[c] = state entering chunk c
__global__ __launch_bounds__(256) void s4_scan_combine(
    const float*  __restrict__ wct,
    const float2* __restrict__ E,
    float2*       __restrict__ I)
{
  int wave = threadIdx.x >> 5, lane = threadIdx.x & 31;
  int bh = blockIdx.x * 8 + wave;              // 0..B*H-1
  int hh = bh % H_;
  float4 p = ((const float4*)wct)[hh * N2_ + lane];
  float pr = p.x, pi = p.y;                    // w
  #pragma unroll
  for (int t = 0; t < 10; ++t) {               // w^(2^10) = w^TCH
    float nr = pr * pr - pi * pi;
    float ni = 2.f * pr * pi;
    pr = nr; pi = ni;
  }
  float ire = 0.f, iim = 0.f;
  for (int c = 0; c < NC_; ++c) {
    I[((size_t)bh * NC_ + c) * N2_ + lane] = make_float2(ire, iim);
    float2 e = E[((size_t)bh * NC_ + c) * N2_ + lane];
    float nre = pr * ire - pi * iim + e.x;     // S_end(c) = w^T * S_end(c-1) + E(c)
    float nim = pr * iim + pi * ire + e.y;
    ire = nre; iim = nim;
  }
}

// Pass 3: re-run chunk with true initial state, emit 2Re(Ct.s)+D*u, GELU, bf16
__global__ __launch_bounds__(256) void s4_conv_out(
    const float*  __restrict__ h,
    const float*  __restrict__ wct,
    const float*  __restrict__ Dp,   // layer slice (H)
    const float2* __restrict__ I,
    bf16*         __restrict__ yg)   // (B,H,L) bf16
{
  int wave = threadIdx.x >> 5, lane = threadIdx.x & 31;
  int gidx = blockIdx.x * 8 + wave;            // 0..B*H*NC-1
  int c  = gidx % NC_;
  int bh = gidx / NC_;
  int b = bh / H_, hh = bh % H_;
  float4 p = ((const float4*)wct)[hh * N2_ + lane];
  float wre = p.x, wim = p.y, cre = p.z, cim = p.w;
  float Dh = Dp[hh];
  const float* hrow = h  + ((size_t)b * H_ + hh) * L_ + (size_t)c * TCH;
  bf16*        orow = yg + ((size_t)b * H_ + hh) * L_ + (size_t)c * TCH;
  float2 s0 = I[((size_t)bh * NC_ + c) * N2_ + lane];
  float sre = s0.x, sim = s0.y;
  for (int l0 = 0; l0 < TCH; l0 += 32) {
    float uvec = hrow[l0 + lane];
    float outv = 0.f;
    #pragma unroll 4
    for (int j = 0; j < 32; ++j) {
      float u = __shfl(uvec, j, 32);
      float nre = wre * sre - wim * sim + u;
      float nim = wre * sim + wim * sre;
      sre = nre; sim = nim;
      float contrib = cre * sre - cim * sim;
      #pragma unroll
      for (int m = 16; m >= 1; m >>= 1) contrib += __shfl_xor(contrib, m, 32);
      float y = 2.0f * contrib + Dh * u;
      if (lane == j) outv = gelu_tanh(y);
    }
    orow[l0 + lane] = (bf16)outv;
  }
}

// ---------------------------------------------------------------------------
// 5) Output GEMM: u[b][m][l] = sum_k Wo[m][k]*yg[b][k][l] + bo[m]  (M=512,K=256)
//    B tile (256 k x 64 cols, bf16) staged by TDM, transposed in LDS.
// ---------------------------------------------------------------------------
__global__ __launch_bounds__(256) void s4_out_gemm(
    const bf16*  __restrict__ ygb,   // (B,H,L) bf16
    const bf16*  __restrict__ Wob,   // layer slice (2H,H) bf16
    const float* __restrict__ bo,    // layer slice (2H)
    float*       __restrict__ u)     // (B,2H,L)
{
  const int COLS = 64;
  __shared__ alignas(16) bf16 raw[H_][COLS];   // [k][col] : TDM tile layout (32 KB)
  __shared__ alignas(32) bf16 yt[COLS][272];   // [col][k] : WMMA B-frag layout
  int tid  = threadIdx.x;
  int col0 = blockIdx.x * COLS;
  int b    = col0 / L_;
  int l0   = col0 % L_;
  const bf16* gsrc = ygb + ((size_t)b * H_) * L_ + l0;  // row k at gsrc + k*L

#if HAVE_TDM
  if (tid < 32)
    tdm_load_2d((unsigned)(size_t)&raw[0][0], gsrc,
                /*2B*/1, L_, H_, COLS, H_, L_);
#else
  for (int i = tid; i < COLS * H_; i += 256) {
    int k = i >> 6, c = i & 63;
    raw[k][c] = gsrc[(size_t)k * L_ + c];
  }
#endif
  __syncthreads();

  // LDS transpose raw[k][c] -> yt[c][k] (k becomes contiguous per column)
  for (int i = tid; i < COLS * H_ / 8; i += 256) {
    int kb = i >> 6;                 // 0..31 (block of 8 k)
    int c  = i & 63;
    bf16 tmp[8];
    #pragma unroll
    for (int j = 0; j < 8; ++j) tmp[j] = raw[kb * 8 + j][c];
    *(v8bf*)&yt[c][kb * 8] = *(const v8bf*)tmp;
  }
  __syncthreads();

  int wave = tid >> 5, lane = tid & 31;
  int hf = lane >> 4, l16 = lane & 15;
  v8f acc[4][4] = {};                          // wave owns 64 rows x 64 cols
  for (int kk = 0; kk < H_; kk += 32) {
    v16bf Bf[4];
    for (int ct = 0; ct < 4; ++ct)
      Bf[ct] = *(const v16bf*)&yt[ct * 16 + l16][kk + 16 * hf];
    for (int rt = 0; rt < 4; ++rt) {
      int m = wave * 64 + rt * 16 + l16;
      const bf16* Ar = Wob + (size_t)m * H_ + kk;
      __builtin_prefetch(Ar + 32, 0, 1);       // -> global_prefetch_b8
      v8bf a0 = *(const v8bf*)(Ar + hf * 16);
      v8bf a1 = *(const v8bf*)(Ar + hf * 16 + 8);
      v16bf A;
      #pragma unroll
      for (int j = 0; j < 8; ++j) { A[j] = a0[j]; A[j + 8] = a1[j]; }
      for (int ct = 0; ct < 4; ++ct)
        acc[rt][ct] = __builtin_amdgcn_wmma_f32_16x16x32_bf16(
            false, A, false, Bf[ct], (short)0, acc[rt][ct], false, false);
    }
  }
  for (int rt = 0; rt < 4; ++rt)
    for (int ct = 0; ct < 4; ++ct)
      #pragma unroll
      for (int v = 0; v < 8; ++v) {
        int row = wave * 64 + rt * 16 + v + 8 * hf;
        int col = ct * 16 + l16;
        u[((size_t)b * (2 * H_) + row) * L_ + l0 + col] = acc[rt][ct][v] + bo[row];
      }
}

// ---------------------------------------------------------------------------
// 6) GLU + residual + LayerNorm over channels, 32-column slab per block
// ---------------------------------------------------------------------------
__global__ __launch_bounds__(256) void glu_ln(
    const float* __restrict__ u,     // (B,2H,L)
    const float* __restrict__ lnw,   // layer slice (H)
    const float* __restrict__ lnb,   // layer slice (H)
    float*       __restrict__ h)     // (B,H,L) updated in place
{
  const int CW = 32;
  __shared__ float zt[H_][CW];
  __shared__ float psum[8][CW], psq[8][CW];
  __shared__ float mu[CW], rstd[CW];
  int tid = threadIdx.x;
  int c = tid & 31, g = tid >> 5;
  int col0 = blockIdx.x * CW;
  int b = col0 / L_, l0 = col0 % L_;
  float s = 0.f, s2 = 0.f;
  for (int ch = g; ch < H_; ch += 8) {
    float a  = u[((size_t)b * 2 * H_ + ch) * L_ + l0 + c];
    float sg = u[((size_t)b * 2 * H_ + ch + H_) * L_ + l0 + c];
    float hp = h[((size_t)b * H_ + ch) * L_ + l0 + c];
    float z = a * (1.0f / (1.0f + __expf(-sg))) + hp;   // GLU + residual
    zt[ch][c] = z;
    s += z; s2 += z * z;
  }
  psum[g][c] = s; psq[g][c] = s2;
  __syncthreads();
  if (g == 0) {
    float ts = 0.f, tq = 0.f;
    #pragma unroll
    for (int k = 0; k < 8; ++k) { ts += psum[k][c]; tq += psq[k][c]; }
    float m = ts * (1.0f / H_);
    float var = tq * (1.0f / H_) - m * m;
    mu[c] = m; rstd[c] = rsqrtf(var + 1e-5f);
  }
  __syncthreads();
  for (int ch = g; ch < H_; ch += 8) {
    float hn = (zt[ch][c] - mu[c]) * rstd[c] * lnw[ch] + lnb[ch];
    h[((size_t)b * H_ + ch) * L_ + l0 + c] = hn;
  }
}

// ---------------------------------------------------------------------------
// 7) Decoder: out[b][l][o] = sum_ch h[b][ch][l]*out_W[o][ch] + out_b[o]
// ---------------------------------------------------------------------------
__global__ __launch_bounds__(256) void decoder(
    const float* __restrict__ h,     // (B,H,L)
    const float* __restrict__ Wd,    // (DOUT,H)
    const float* __restrict__ bd,    // (DOUT)
    float*       __restrict__ out)   // (B,L,DOUT)
{
  __shared__ float Ws[DOUT][H_];
  int tid = threadIdx.x;
  for (int i = tid; i < DOUT * H_; i += 256) Ws[i / H_][i % H_] = Wd[i];
  __syncthreads();
  int col = blockIdx.x * 256 + tid;            // b*L + l ; L multiple of 256
  int b = col / L_, l = col % L_;
  float acc[DOUT];
  #pragma unroll
  for (int o = 0; o < DOUT; ++o) acc[o] = bd[o];
  for (int ch = 0; ch < H_; ++ch) {
    float hv = h[((size_t)b * H_ + ch) * L_ + l];
    #pragma unroll
    for (int o = 0; o < DOUT; ++o) acc[o] += hv * Ws[o][ch];
  }
  #pragma unroll
  for (int o = 0; o < DOUT; ++o) out[(size_t)col * DOUT + o] = acc[o];
}

// ---------------------------------------------------------------------------
// Launch
// ---------------------------------------------------------------------------
extern "C" void kernel_launch(void* const* d_in, const int* in_sizes, int n_in,
                              void* d_out, int out_size, void* d_ws, size_t ws_size,
                              hipStream_t stream) {
  const float* x      = (const float*)d_in[0];
  const float* enc_W  = (const float*)d_in[1];
  const float* enc_b  = (const float*)d_in[2];
  const float* log_dt = (const float*)d_in[3];
  const float* log_A  = (const float*)d_in[4];
  const float* A_im   = (const float*)d_in[5];
  const float* C_re   = (const float*)d_in[6];
  const float* C_im   = (const float*)d_in[7];
  const float* Dp     = (const float*)d_in[8];
  const float* Wo     = (const float*)d_in[9];
  const float* bo     = (const float*)d_in[10];
  const float* lnw    = (const float*)d_in[11];
  const float* lnb    = (const float*)d_in[12];
  const float* out_W  = (const float*)d_in[13];
  const float* out_b  = (const float*)d_in[14];
  float* out = (float*)d_out;

  char* ws = (char*)d_ws;
  size_t off = 0;
  float*  hbuf = (float*)(ws + off);  off += (size_t)B_ * H_ * L_ * 4;       // 32 MB
  bf16*   ygb  = (bf16*)(ws + off);   off += (size_t)B_ * H_ * L_ * 2;       // 16 MB
  float*  ubuf = (float*)(ws + off);  off += (size_t)B_ * 2 * H_ * L_ * 4;   // 64 MB
  float*  wct  = (float*)(ws + off);  off += (size_t)NL_ * H_ * N2_ * 16;    // 512 KB
  bf16*   Wob  = (bf16*)(ws + off);   off += (size_t)NL_ * 2 * H_ * H_ * 2;  // 1 MB
  bf16*   eWb  = (bf16*)(ws + off);   off += (size_t)H_ * DIN * 2;
  float2* Ebuf = (float2*)(ws + off); off += (size_t)B_ * H_ * NC_ * N2_ * 8; // 2 MB
  float2* Ibuf = (float2*)(ws + off); off += (size_t)B_ * H_ * NC_ * N2_ * 8; // 2 MB

  s4d_params<<<(NL_ * H_ * N2_ + 255) / 256, 256, 0, stream>>>(
      log_dt, log_A, A_im, C_re, C_im, wct);
  f32_to_bf16<<<(NL_ * 2 * H_ * H_ + 255) / 256, 256, 0, stream>>>(
      Wo, Wob, NL_ * 2 * H_ * H_);
  f32_to_bf16<<<(H_ * DIN + 255) / 256, 256, 0, stream>>>(enc_W, eWb, H_ * DIN);

  encoder_gemm<<<B_ * L_ / 64, 256, 0, stream>>>(x, eWb, enc_b, hbuf);

  for (int i = 0; i < NL_; ++i) {
    const float* wct_i = wct + (size_t)i * H_ * N2_ * 4;
    s4_scan_local<<<B_ * H_ * NC_ / 8, 256, 0, stream>>>(hbuf, wct_i, Ebuf);
    s4_scan_combine<<<B_ * H_ / 8, 256, 0, stream>>>(wct_i, Ebuf, Ibuf);
    s4_conv_out<<<B_ * H_ * NC_ / 8, 256, 0, stream>>>(
        hbuf, wct_i, Dp + i * H_, Ibuf, ygb);
    s4_out_gemm<<<B_ * L_ / 64, 256, 0, stream>>>(
        ygb, Wob + (size_t)i * 2 * H_ * H_, bo + i * 2 * H_, ubuf);
    glu_ln<<<B_ * L_ / 32, 256, 0, stream>>>(
        ubuf, lnw + i * H_, lnb + i * H_, hbuf);
  }

  decoder<<<B_ * L_ / 256, 256, 0, stream>>>(hbuf, out_W, out_b, out);
}
